// DeepSetAttentionModel_34978213659076
// MI455X (gfx1250) — matrix operand order repaired and checked by reading, not compiled, not run on hardware
//
#include <hip/hip_runtime.h>
#include <hip/hip_bf16.h>
#include <math.h>

typedef _Float16 v16h __attribute__((ext_vector_type(16)));
typedef _Float16 h8   __attribute__((ext_vector_type(8)));
typedef float    v8f  __attribute__((ext_vector_type(8)));

#define T_OBS 4096
#define TP1   4097          // T+1 rows per batch (demo row prepended)
#define NB    64
#define NH    4
#define LAT   128

// workspace layout (bytes)
#define WS_DEMO   0            // demo_enc  [64][32] f32      (8192 B)
#define WS_WEFF   8192         // w_eff     [32][4]  f32      (512 B)
#define WS_STATS  8704         // stats     [256][2] f32      (2048 B)
#define WS_FRAGS  10752        // 72 frags x 512 f16          (73728 B)
#define WS_FEAT   84480        // feat      [64][512] f32     (131072 B)
#define WS_PRE    215552       // preattn   [64][4][4097] f32 (4195328 B)
#define WS_ENC    4410880      // enc       [64][4097][128] f32 (134250496 B)

// ---------------------------------------------------------------------------
// prep: demo MLP, folded attention vector, f16 weight fragment swizzle, zero feat
// ---------------------------------------------------------------------------
__global__ __launch_bounds__(256) void prep_kernel(
    const float* __restrict__ demo,
    const float* __restrict__ demo_w1, const float* __restrict__ demo_b1,
    const float* __restrict__ demo_w2, const float* __restrict__ demo_b2,
    const float* __restrict__ phi_w1, const float* __restrict__ phi_w2,
    const float* __restrict__ phi_w3,
    const float* __restrict__ W_k, const float* __restrict__ W_q,
    float* __restrict__ demo_enc, float* __restrict__ weff,
    _Float16* __restrict__ frags, float* __restrict__ feat)
{
    const int blk = blockIdx.x;
    const int tid = threadIdx.x;
    if (blk == 0) {
        // demo encoder: relu(demo@w1+b1)@w2+b2  -> [64][32]
        __shared__ float hid[64][128];
        for (int idx = tid; idx < 64 * 128; idx += 256) {
            int row = idx >> 7, j = idx & 127;
            float a = demo_b1[j];
            for (int k = 0; k < 8; ++k) a += demo[row * 8 + k] * demo_w1[k * 128 + j];
            hid[row][j] = fmaxf(a, 0.f);
        }
        __syncthreads();
        for (int idx = tid; idx < 64 * 32; idx += 256) {
            int row = idx >> 5, c = idx & 31;
            float a = demo_b2[c];
            for (int k = 0; k < 128; ++k) a += hid[row][k] * demo_w2[k * 32 + c];
            demo_enc[row * 32 + c] = a;
        }
    } else if (blk == 1) {
        // w_eff[c][h] = sum_d W_k[c, h*64+d] * W_q[h,d] / sqrt(64)
        if (tid < 128) {
            int c = tid >> 2, h = tid & 3;
            float a = 0.f;
            for (int d = 0; d < 64; ++d) a += W_k[c * 256 + h * 64 + d] * W_q[h * 64 + d];
            weff[c * 4 + h] = a * 0.125f;
        }
    } else if (blk == 2) {
        for (int idx = tid; idx < 64 * 512; idx += 256) feat[idx] = 0.f;
    } else if (blk >= 4 && blk < 76) {
        // f16 B-matrix fragments in WMMA lane layout:
    // element e=(lane,j): n = nt*16 + (lane&15), k = ks*32 + ((lane&16)>>1) + j + (j&8)
        int f = blk - 4;
        const float* W; int nt, ks;
        if (f < 8)       { W = phi_w1; nt = f;              ks = 0; }
        else if (f < 40) { int i2 = f - 8;  W = phi_w2; nt = i2 >> 2; ks = i2 & 3; }
        else             { int i2 = f - 40; W = phi_w3; nt = i2 >> 2; ks = i2 & 3; }
        for (int e = tid; e < 512; e += 256) {
            int ln = e >> 4, j = e & 15;
            int n = nt * 16 + (ln & 15);
            int k = ks * 32 + ((ln & 16) >> 1) + j + (j & 8);
            frags[(size_t)f * 512 + e] = (_Float16)W[k * 128 + n];
        }
    }
}

// ---------------------------------------------------------------------------
// phase1: build x, preattn (= x @ w_eff, masked), phi MLP via WMMA -> enc
// grid (65, 64), block 128 (4 waves, 16 rows/wave)
// ---------------------------------------------------------------------------
__global__ __launch_bounds__(128) void phase1_kernel(
    const float* __restrict__ times, const float* __restrict__ values,
    const int* __restrict__ meas, const int* __restrict__ lengths,
    const float* __restrict__ demo_enc, const float* __restrict__ weff,
    const _Float16* __restrict__ frags,
    const float* __restrict__ phi_b1, const float* __restrict__ phi_b2,
    const float* __restrict__ phi_b3,
    float* __restrict__ preattn, float* __restrict__ enc)
{
    __shared__ __align__(16) float     xs[4][16][33];     // padded: conflict-free
    __shared__ __align__(16) _Float16  act1[4][16][136];  // padded rows (272B)
    __shared__ __align__(16) _Float16  act2[4][16][136];

    const int b    = blockIdx.y;
    const int wave = threadIdx.x >> 5;
    const int lane = threadIdx.x & 31;
    const int t0   = blockIdx.x * 64 + wave * 16;
    const int len  = lengths[b];

    // ---- build input features x[16][32] for this tile ----
    for (int idx = lane; idx < 16 * 32; idx += 32) {
        int r = idx >> 5, c = idx & 31;
        int t = t0 + r;
        float v = 0.f;
        if (t == 0) {
            v = demo_enc[b * 32 + c];
        } else if (t <= T_OBS) {
            int ti = t - 1;
            if (c < 16) {
                int k = c & 7;                         // timescale index
                float invts = __expf(-0.6578814552f * (float)k); // 100^(-k/7)
                float arg = times[b * T_OBS + ti] * invts;
                v = (c < 8) ? __sinf(arg) : __cosf(arg);
            } else if (c == 16) {
                v = values[b * T_OBS + ti];
            } else {
                v = (meas[b * T_OBS + ti] == (c - 17)) ? 1.f : 0.f;
            }
        }
        xs[wave][r][c] = v;
    }
    __syncthreads();

    // ---- preattn: x @ w_eff (W_q folded into W_k; agg term cancels in softmax) ----
    if (lane < 16) {
        int t = t0 + lane;
        if (t < TP1) {
            float acc[4] = {0.f, 0.f, 0.f, 0.f};
            for (int c = 0; c < 32; ++c) {
                float xv = xs[wave][lane][c];
                #pragma unroll
                for (int h = 0; h < 4; ++h) acc[h] += xv * weff[c * 4 + h];
            }
            bool ok = (t <= len);   // mask: t < length+1
            #pragma unroll
            for (int h = 0; h < 4; ++h)
                preattn[((size_t)b * 4 + h) * TP1 + t] = ok ? acc[h] : -1e9f;
        }
    }

    const int m16  = lane & 15;
    const int Lofs = (lane & 16) >> 1;   // 0 or 8

    // ---- layer 1: A = x (16x32 f16), B frags 0..7 ----
    v16h aX;
    #pragma unroll
    for (int j = 0; j < 16; ++j) {
        int k = j + (j & 8) + Lofs;      // 16-bit A-matrix lane layout
        aX[j] = (_Float16)xs[wave][m16][k];
    }
    const _Float16* fr1 = frags;
    const _Float16* fr2 = frags + 8  * 512;
    const _Float16* fr3 = frags + 40 * 512;

    #pragma unroll
    for (int nt = 0; nt < 8; ++nt) {
        int n0 = nt * 16 + m16;
        float bn = phi_b1[n0];
        v8f c = {bn, bn, bn, bn, bn, bn, bn, bn};
        v16h bf = *(const v16h*)(fr1 + (size_t)nt * 512 + lane * 16);
        c = __builtin_amdgcn_wmma_f32_16x16x32_f16(false, aX, false, bf,
                                                   (short)0, c, false, false);
        #pragma unroll
        for (int r = 0; r < 8; ++r) {
            float v = c[r] > 0.f ? c[r] : 0.f;
            act1[wave][r + Lofs][n0] = (_Float16)v;
        }
    }
    __syncthreads();

    // ---- layer 2: K=128 (4 ksteps), act1 -> act2 ----
    v16h a2[4];
    #pragma unroll
    for (int ks = 0; ks < 4; ++ks) {
        h8 lo = *(const h8*)(&act1[wave][m16][ks * 32 + Lofs]);
        h8 hi = *(const h8*)(&act1[wave][m16][ks * 32 + 16 + Lofs]);
        a2[ks] = __builtin_shufflevector(lo, hi, 0,1,2,3,4,5,6,7,8,9,10,11,12,13,14,15);
    }
    #pragma unroll
    for (int nt = 0; nt < 8; ++nt) {
        int n0 = nt * 16 + m16;
        float bn = phi_b2[n0];
        v8f c = {bn, bn, bn, bn, bn, bn, bn, bn};
        #pragma unroll
        for (int ks = 0; ks < 4; ++ks) {
            v16h bf = *(const v16h*)(fr2 + (size_t)(nt * 4 + ks) * 512 + lane * 16);
            c = __builtin_amdgcn_wmma_f32_16x16x32_f16(false, a2[ks], false, bf,
                                                       (short)0, c, false, false);
        }
        #pragma unroll
        for (int r = 0; r < 8; ++r) {
            float v = c[r] > 0.f ? c[r] : 0.f;
            act2[wave][r + Lofs][n0] = (_Float16)v;
        }
    }
    __syncthreads();

    // ---- layer 3: act2 -> enc (global, f32) ----
    v16h a3[4];
    #pragma unroll
    for (int ks = 0; ks < 4; ++ks) {
        h8 lo = *(const h8*)(&act2[wave][m16][ks * 32 + Lofs]);
        h8 hi = *(const h8*)(&act2[wave][m16][ks * 32 + 16 + Lofs]);
        a3[ks] = __builtin_shufflevector(lo, hi, 0,1,2,3,4,5,6,7,8,9,10,11,12,13,14,15);
    }
    #pragma unroll
    for (int nt = 0; nt < 8; ++nt) {
        int n0 = nt * 16 + m16;
        float bn = phi_b3[n0];
        v8f c = {bn, bn, bn, bn, bn, bn, bn, bn};
        #pragma unroll
        for (int ks = 0; ks < 4; ++ks) {
            v16h bf = *(const v16h*)(fr3 + (size_t)(nt * 4 + ks) * 512 + lane * 16);
            c = __builtin_amdgcn_wmma_f32_16x16x32_f16(false, a3[ks], false, bf,
                                                       (short)0, c, false, false);
        }
        #pragma unroll
        for (int r = 0; r < 8; ++r) {
            int t = t0 + r + Lofs;
            if (t < TP1) {
                float v = c[r] > 0.f ? c[r] : 0.f;
                enc[((size_t)b * TP1 + t) * LAT + n0] = v;
            }
        }
    }
}

// ---------------------------------------------------------------------------
// phase2: per (b,h) online max / sum-exp over 4097 preattn values
// ---------------------------------------------------------------------------
__global__ __launch_bounds__(256) void phase2_kernel(
    const float* __restrict__ preattn, float* __restrict__ stats)
{
    __shared__ float sm[256], ss[256];
    const int bh  = blockIdx.x;
    const int tid = threadIdx.x;
    const float* p = preattn + (size_t)bh * TP1;
    float m = -INFINITY, s = 0.f;
    for (int i = tid; i < TP1; i += 256) {
        float v = p[i];
        float nm = fmaxf(m, v);
        s = s * __expf(m - nm) + __expf(v - nm);
        m = nm;
    }
    sm[tid] = m; ss[tid] = s;
    __syncthreads();
    for (int st = 128; st > 0; st >>= 1) {
        if (tid < st) {
            float m2 = sm[tid + st], s2 = ss[tid + st];
            float nm = fmaxf(sm[tid], m2);
            ss[tid] = ss[tid] * __expf(sm[tid] - nm) + s2 * __expf(m2 - nm);
            sm[tid] = nm;
        }
        __syncthreads();
    }
    if (tid == 0) { stats[bh * 2] = sm[0]; stats[bh * 2 + 1] = ss[0]; }
}

// ---------------------------------------------------------------------------
// phase3: head_agg accumulation: feat[b][h*128+c] += sum_t attn(b,t,h)*enc(b,t,c)
// grid (17, 64), block 128: thread = column c, 4 head accumulators
// ---------------------------------------------------------------------------
__global__ __launch_bounds__(128) void phase3_kernel(
    const float* __restrict__ preattn, const float* __restrict__ stats,
    const float* __restrict__ enc, float* __restrict__ feat)
{
    __shared__ float wr[4][256];
    const int b     = blockIdx.y;
    const int tbase = blockIdx.x * 256;
    int rmax = TP1 - tbase; if (rmax > 256) rmax = 256;

    float mh[4], is[4];
    #pragma unroll
    for (int h = 0; h < 4; ++h) {
        mh[h] = stats[(b * 4 + h) * 2];
        is[h] = 1.f / stats[(b * 4 + h) * 2 + 1];
    }
    for (int idx = threadIdx.x; idx < 4 * 256; idx += 128) {
        int h = idx >> 8, r = idx & 255;
        float w = 0.f;
        if (r < rmax)
            w = __expf(preattn[((size_t)b * 4 + h) * TP1 + tbase + r] - mh[h]) * is[h];
        wr[h][r] = w;
    }
    __syncthreads();

    const int c = threadIdx.x;
    float a0 = 0.f, a1 = 0.f, a2 = 0.f, a3 = 0.f;
    for (int r = 0; r < rmax; ++r) {
        float e = enc[((size_t)b * TP1 + tbase + r) * LAT + c];
        a0 += wr[0][r] * e; a1 += wr[1][r] * e;
        a2 += wr[2][r] * e; a3 += wr[3][r] * e;
    }
    atomicAdd(&feat[b * 512 + 0 * 128 + c], a0);
    atomicAdd(&feat[b * 512 + 1 * 128 + c], a1);
    atomicAdd(&feat[b * 512 + 2 * 128 + c], a2);
    atomicAdd(&feat[b * 512 + 3 * 128 + c], a3);
}

// ---------------------------------------------------------------------------
// phase4: rho MLP + sigmoid (tiny: 64x512 -> 128 -> 128 -> 1)
// ---------------------------------------------------------------------------
__global__ __launch_bounds__(256) void phase4_kernel(
    const float* __restrict__ feat,
    const float* __restrict__ w1, const float* __restrict__ b1,
    const float* __restrict__ w2, const float* __restrict__ b2,
    const float* __restrict__ w3, const float* __restrict__ b3,
    float* __restrict__ out)
{
    __shared__ float h1[64][128];
    __shared__ float h2[64][128];
    const int tid = threadIdx.x;
    for (int idx = tid; idx < 64 * 128; idx += 256) {
        int row = idx >> 7, j = idx & 127;
        float a = b1[j];
        for (int k = 0; k < 512; ++k) a += feat[row * 512 + k] * w1[k * 128 + j];
        h1[row][j] = fmaxf(a, 0.f);
    }
    __syncthreads();
    for (int idx = tid; idx < 64 * 128; idx += 256) {
        int row = idx >> 7, j = idx & 127;
        float a = b2[j];
        for (int k = 0; k < 128; ++k) a += h1[row][k] * w2[k * 128 + j];
        h2[row][j] = fmaxf(a, 0.f);
    }
    __syncthreads();
    if (tid < 64) {
        float a = b3[0];
        for (int k = 0; k < 128; ++k) a += h2[tid][k] * w3[k];
        out[tid] = 1.f / (1.f + __expf(-a));
    }
}

// ---------------------------------------------------------------------------
extern "C" void kernel_launch(void* const* d_in, const int* in_sizes, int n_in,
                              void* d_out, int out_size, void* d_ws, size_t ws_size,
                              hipStream_t stream) {
    (void)in_sizes; (void)n_in; (void)out_size; (void)ws_size;
    const float* demo     = (const float*)d_in[0];
    const float* times    = (const float*)d_in[1];
    const float* values   = (const float*)d_in[2];
    const int*   meas     = (const int*)d_in[3];
    const int*   lengths  = (const int*)d_in[4];
    const float* demo_w1  = (const float*)d_in[5];
    const float* demo_b1  = (const float*)d_in[6];
    const float* demo_w2  = (const float*)d_in[7];
    const float* demo_b2  = (const float*)d_in[8];
    const float* phi_w1   = (const float*)d_in[9];
    const float* phi_b1   = (const float*)d_in[10];
    const float* phi_w2   = (const float*)d_in[11];
    const float* phi_b2   = (const float*)d_in[12];
    const float* phi_w3   = (const float*)d_in[13];
    const float* phi_b3   = (const float*)d_in[14];
    // d_in[15..22]: psi_* / rho_attn_* — mathematically dead (softmax shift-invariance)
    const float* W_k      = (const float*)d_in[23];
    const float* W_q      = (const float*)d_in[24];
    const float* rho_w1   = (const float*)d_in[25];
    const float* rho_b1   = (const float*)d_in[26];
    const float* rho_w2   = (const float*)d_in[27];
    const float* rho_b2   = (const float*)d_in[28];
    const float* rho_w3   = (const float*)d_in[29];
    const float* rho_b3   = (const float*)d_in[30];

    char* ws = (char*)d_ws;
    float*     demo_enc = (float*)(ws + WS_DEMO);
    float*     weff     = (float*)(ws + WS_WEFF);
    float*     stats    = (float*)(ws + WS_STATS);
    _Float16*  frags    = (_Float16*)(ws + WS_FRAGS);
    float*     feat     = (float*)(ws + WS_FEAT);
    float*     preattn  = (float*)(ws + WS_PRE);
    float*     enc      = (float*)(ws + WS_ENC);

    prep_kernel<<<76, 256, 0, stream>>>(demo, demo_w1, demo_b1, demo_w2, demo_b2,
                                        phi_w1, phi_w2, phi_w3, W_k, W_q,
                                        demo_enc, weff, frags, feat);
    phase1_kernel<<<dim3(65, 64), 128, 0, stream>>>(times, values, meas, lengths,
                                                    demo_enc, weff, frags,
                                                    phi_b1, phi_b2, phi_b3,
                                                    preattn, enc);
    phase2_kernel<<<256, 256, 0, stream>>>(preattn, stats);
    phase3_kernel<<<dim3(17, 64), 128, 0, stream>>>(preattn, stats, enc, feat);
    phase4_kernel<<<1, 256, 0, stream>>>(feat, rho_w1, rho_b1, rho_w2, rho_b2,
                                         rho_w3, rho_b3, (float*)d_out);
}
